// RecurrentRelationalNetwork_36464272343752
// MI455X (gfx1250) — matrix operand order repaired and checked by reading, not compiled, not run on hardware
//
#include <hip/hip_runtime.h>

#define N_NODES 82944
#define N_EDGES 1658880
#define HID     96
#define NSTEPS  16

// ---------------------------------------------------------------- WMMA helpers
typedef __attribute__((ext_vector_type(16))) __bf16 v16bf;
typedef __attribute__((ext_vector_type(8)))  float  v8f;

union FragBF { v16bf v; uint4 q[2]; };

__device__ __forceinline__ unsigned short f2bf(float f) {
  unsigned int u = __float_as_uint(f);
  u += 0x7FFFu + ((u >> 16) & 1u);           // round-to-nearest-even
  return (unsigned short)(u >> 16);
}

// A-matrix 16x32 bf16 fragment from row-major LDS [16 x lda] (ISA 7.12.2):
// per lane: rows of 8 consecutive bf16 at k0+kb and k0+16+kb  -> 2x ds_load_b128
__device__ __forceinline__ v16bf lda_frag(const unsigned short* A, int lda, int k0) {
  const int lane = threadIdx.x & 31;
  const int m  = lane & 15;
  const int kb = (lane >> 4) << 3;           // 0 or 8
  const unsigned short* base = A + m * lda + k0 + kb;
  FragBF f;
  f.q[0] = *(const uint4*)(base);
  f.q[1] = *(const uint4*)(base + 16);
  return f.v;
}

// B fragment from pre-swizzled weights: tile-major, lane-major, 16 bf16/lane
__device__ __forceinline__ v16bf ldb_frag_sw(const unsigned short* Bs, int tile) {
  const int lane = threadIdx.x & 31;
  const uint4* p = (const uint4*)(Bs + ((tile * 32 + lane) << 4));
  FragBF f;
  f.q[0] = p[0];
  f.q[1] = p[1];
  return f.v;
}

#define WMMA_BF16(a, b, c) \
  __builtin_amdgcn_wmma_f32_16x16x32_bf16(false, (a), false, (b), (short)0, (c), false, false)

// ---------------------------------------------------------------- async LDS copy
#if defined(__has_builtin)
#  if __has_builtin(__builtin_amdgcn_global_load_async_to_lds_b128)
#    define HAVE_ASYNC_LDS 1
#  endif
#endif
#ifndef HAVE_ASYNC_LDS
#  define HAVE_ASYNC_LDS 0
#endif

#if HAVE_ASYNC_LDS
// param types per hipcc diagnostic: (int4 __device__*, int4 __shared__*, imm, imm)
typedef int v4i_async __attribute__((vector_size(4 * sizeof(int))));
__device__ __forceinline__ void async_b128(void* lds_dst, const void* gsrc) {
  __builtin_amdgcn_global_load_async_to_lds_b128(
      (__attribute__((address_space(1))) v4i_async*)gsrc,
      (__attribute__((address_space(3))) v4i_async*)lds_dst, 0, 0);
}
__device__ __forceinline__ void async_wait() {
  asm volatile("s_wait_asynccnt 0" ::: "memory");
}
#endif

// contiguous global -> LDS copy in 16B chunks
__device__ __forceinline__ void copy_to_lds_b128(void* dst, const void* src,
                                                 int nbytes, int tid, int nthreads) {
  const int chunks = nbytes >> 4;
#if HAVE_ASYNC_LDS
  for (int c = tid; c < chunks; c += nthreads)
    async_b128((char*)dst + (c << 4), (const char*)src + (c << 4));
#else
  uint4* d = (uint4*)dst;
  const uint4* s = (const uint4*)src;
  for (int c = tid; c < chunks; c += nthreads) d[c] = s[c];
#endif
}

// ---------------------------------------------------------------- prep kernels
// Swizzle a row-major [K x N] fp32 weight into B-fragment order (bf16):
// element i -> tile = i>>9 (kt*NT+nt), lane = (i>>4)&31, j = i&15
// k = kt*32 + ((lane>>4)<<3) + (j&7) + ((j>>3)<<4) ; n = nt*16 + (lane&15) ; zero-pad k >= kvalid
__device__ __forceinline__ unsigned short swz_elem(const float* W, int N, int kvalid,
                                                   int NT, int i) {
  const int j = i & 15;
  const int lane = (i >> 4) & 31;
  const int tile = i >> 9;
  const int kt = tile / NT, nt = tile - kt * NT;
  const int k = kt * 32 + ((lane >> 4) << 3) + (j & 7) + ((j >> 3) << 4);
  const int n = nt * 16 + (lane & 15);
  return (k < kvalid) ? f2bf(W[k * N + n]) : (unsigned short)0;
}

__global__ void prep_weights_kernel(const float* __restrict__ W1,
                                    const float* __restrict__ W2,
                                    const float* __restrict__ W3,
                                    const float* __restrict__ gk,
                                    const float* __restrict__ grk,
                                    unsigned short* __restrict__ W1s,
                                    unsigned short* __restrict__ W2s,
                                    unsigned short* __restrict__ W3s,
                                    unsigned short* __restrict__ Ks,
                                    unsigned short* __restrict__ Rs) {
  const int i = blockIdx.x * blockDim.x + threadIdx.x;
  if (i < 36 * 512) W1s[i] = swz_elem(W1, 96, 192, 6, i);      // K=192, 6x6 tiles
  if (i < 18 * 512) { W2s[i] = swz_elem(W2, 96, 96, 6, i);     // K=96, 3x6 tiles
                      W3s[i] = swz_elem(W3, 96, 96, 6, i); }
  if (i < 72 * 512) Ks[i]  = swz_elem(gk, 288, 106, 18, i);    // K=128(pad), 4x18
  if (i < 54 * 512) Rs[i]  = swz_elem(grk, 288, 96, 18, i);    // K=96, 3x18
}

__global__ void init_h_kernel(const float* __restrict__ h0,
                              float* __restrict__ hf,
                              unsigned short* __restrict__ hb, int n) {
  const int i = blockIdx.x * blockDim.x + threadIdx.x;
  if (i < n) { const float v = h0[i]; hf[i] = v; hb[i] = f2bf(v); }
}

// ---------------------------------------------------------------- edge MLP
// 256 threads = 8 waves; 16 edges per wave, 128 edges per workgroup.
__global__ __launch_bounds__(256)
void edge_mlp_kernel(const unsigned short* __restrict__ hb,
                     const int* __restrict__ esrc,
                     const int* __restrict__ etgt,
                     const unsigned short* __restrict__ W1s,
                     const unsigned short* __restrict__ W2s,
                     const unsigned short* __restrict__ W3s,
                     const float* __restrict__ b1,
                     const float* __restrict__ b2,
                     const float* __restrict__ b3,
                     float* __restrict__ msgs) {
  extern __shared__ unsigned short smem_us[];
  unsigned short* sW1 = smem_us;              // 36*512
  unsigned short* sW2 = sW1 + 36 * 512;       // 18*512
  unsigned short* sW3 = sW2 + 18 * 512;       // 18*512
  unsigned short* sA  = sW3 + 18 * 512;

  const int tid  = threadIdx.x;
  const int wave = tid >> 5;
  const int lane = tid & 31;

  copy_to_lds_b128(sW1, W1s, 36 * 1024, tid, 256);
  copy_to_lds_b128(sW2, W2s, 18 * 1024, tid, 256);
  copy_to_lds_b128(sW3, W3s, 18 * 1024, tid, 256);

  unsigned short* X = sA + wave * (16 * 192 + 16 * 96);
  unsigned short* Y = X + 16 * 192;

  const int e0 = (blockIdx.x * 8 + wave) * 16;
  const int tg = etgt[e0 + (lane & 15)];
  const int sr = esrc[e0 + (lane & 15)];

  // gather row m = [ h[target] (96) | h[source] (96) ] bf16; 24 x 16B chunks/row
#if HAVE_ASYNC_LDS
  for (int idx = lane; idx < 16 * 24; idx += 32) {
    const int m = idx / 24;
    const int r = idx - m * 24;                 // 0..11 target, 12..23 source
    const int node = (r < 12) ? __shfl(tg, m, 32) : __shfl(sr, m, 32);
    const int d = (r < 12) ? r : (r - 12);
    async_b128(X + m * 192 + (r << 3), hb + (size_t)node * HID + (d << 3));
  }
  async_wait();
#else
  for (int m = 0; m < 16; ++m) {
    const int t = __shfl(tg, m, 32);
    const int s = __shfl(sr, m, 32);
    const uint4* ht = (const uint4*)(hb + (size_t)t * HID);
    const uint4* hs = (const uint4*)(hb + (size_t)s * HID);
    uint4* xr = (uint4*)(X + m * 192);
    for (int d = lane; d < 12; d += 32) { xr[d] = ht[d]; xr[12 + d] = hs[d]; }
  }
#endif
  __syncthreads();

  // ---- layer 1: [16x192] @ [192x96] + b1, relu -> Y
  v16bf a1[6];
#pragma unroll
  for (int kt = 0; kt < 6; ++kt) a1[kt] = lda_frag(X, 192, kt * 32);
#pragma unroll
  for (int nt = 0; nt < 6; ++nt) {
    v8f c = {};
#pragma unroll
    for (int kt = 0; kt < 6; ++kt) c = WMMA_BF16(a1[kt], ldb_frag_sw(sW1, kt * 6 + nt), c);
    const int n = nt * 16 + (lane & 15);
    const float bias = b1[n];
#pragma unroll
    for (int i = 0; i < 8; ++i) {
      const int m = ((lane >> 4) << 3) + i;
      float v = c[i] + bias;
      Y[m * 96 + n] = f2bf(v > 0.f ? v : 0.f);
    }
  }

  // ---- layer 2: [16x96] @ [96x96] + b2, relu -> X (reused as 16x96)
  v16bf a2[3];
#pragma unroll
  for (int kt = 0; kt < 3; ++kt) a2[kt] = lda_frag(Y, 96, kt * 32);
#pragma unroll
  for (int nt = 0; nt < 6; ++nt) {
    v8f c = {};
#pragma unroll
    for (int kt = 0; kt < 3; ++kt) c = WMMA_BF16(a2[kt], ldb_frag_sw(sW2, kt * 6 + nt), c);
    const int n = nt * 16 + (lane & 15);
    const float bias = b2[n];
#pragma unroll
    for (int i = 0; i < 8; ++i) {
      const int m = ((lane >> 4) << 3) + i;
      float v = c[i] + bias;
      X[m * 96 + n] = f2bf(v > 0.f ? v : 0.f);
    }
  }

  // ---- layer 3: [16x96] @ [96x96] + b3 -> scatter-add into msgs[target]
  v16bf a3[3];
#pragma unroll
  for (int kt = 0; kt < 3; ++kt) a3[kt] = lda_frag(X, 96, kt * 32);
#pragma unroll
  for (int nt = 0; nt < 6; ++nt) {
    v8f c = {};
#pragma unroll
    for (int kt = 0; kt < 3; ++kt) c = WMMA_BF16(a3[kt], ldb_frag_sw(sW3, kt * 6 + nt), c);
    const int n = nt * 16 + (lane & 15);
    const float bias = b3[n];
#pragma unroll
    for (int i = 0; i < 8; ++i) {
      const int m = ((lane >> 4) << 3) + i;
      const int t = __shfl(tg, m, 32);
      unsafeAtomicAdd(&msgs[(size_t)t * HID + n], c[i] + bias);
    }
  }
}

// ---------------------------------------------------------------- fused GRU
// 128 threads = 4 waves; 16 nodes per wave, 64 nodes per workgroup.
__global__ __launch_bounds__(128)
void gru_kernel(const float* __restrict__ clues,
                const float* __restrict__ msgs,
                float* __restrict__ hf,
                unsigned short* __restrict__ hb,
                const unsigned short* __restrict__ Ks,
                const unsigned short* __restrict__ Rs,
                const float* __restrict__ gb /* [2,288] */) {
  extern __shared__ unsigned char smem_raw[];
  unsigned short* sK = (unsigned short*)smem_raw;      // 72*512
  unsigned short* sR = sK + 72 * 512;                  // 54*512
  unsigned char*  pw = (unsigned char*)(sR + 54 * 512);

  const int tid  = threadIdx.x;
  const int wave = tid >> 5;
  const int lane = tid & 31;

  unsigned char* wbase = pw + wave * 31744;
  unsigned short* Ax = (unsigned short*)wbase;         // 16*128 bf16
  unsigned short* Ah = Ax + 16 * 128;                  // 16*96 bf16
  float* Szr = (float*)(Ah + 16 * 96);                 // 16*192 (z,r combined)
  float* Sxh = Szr + 16 * 192;                         // 16*96 (x-side cand)
  float* Shh = Sxh + 16 * 96;                          // 16*96 (h-side cand)

  copy_to_lds_b128(sK, Ks, 72 * 1024, tid, 128);
  copy_to_lds_b128(sR, Rs, 54 * 1024, tid, 128);

  const int node0 = (blockIdx.x * 4 + wave) * 16;

  // stage Ax = [clues | msgs | 0] as packed dword stores
  for (int idx = lane; idx < 16 * 64; idx += 32) {
    const int m = idx >> 6;
    const int k = (idx & 63) * 2;
    const int node = node0 + m;
    float v0 = 0.f, v1 = 0.f;
    if (k < 10)            v0 = clues[node * 10 + k];
    else if (k < 106)      v0 = msgs[(size_t)node * HID + (k - 10)];
    if (k + 1 < 10)        v1 = clues[node * 10 + k + 1];
    else if (k + 1 < 106)  v1 = msgs[(size_t)node * HID + (k + 1 - 10)];
    ((unsigned int*)Ax)[idx] =
        (unsigned int)f2bf(v0) | ((unsigned int)f2bf(v1) << 16);
  }
  // stage Ah = h rows (contiguous bf16)
  copy_to_lds_b128(Ah, hb + (size_t)node0 * HID, 16 * 96 * 2, lane, 32);
#if HAVE_ASYNC_LDS
  async_wait();
#endif
  __syncthreads();

  v16bf ax[4], ah[3];
#pragma unroll
  for (int kt = 0; kt < 4; ++kt) ax[kt] = lda_frag(Ax, 128, kt * 32);
#pragma unroll
  for (int kt = 0; kt < 3; ++kt) ah[kt] = lda_frag(Ah, 96, kt * 32);

#pragma unroll
  for (int j = 0; j < 18; ++j) {
    const int n0 = j * 16;
    v8f cx = {};
#pragma unroll
    for (int kt = 0; kt < 4; ++kt) cx = WMMA_BF16(ax[kt], ldb_frag_sw(sK, kt * 18 + j), cx);
    v8f ch = {};
#pragma unroll
    for (int kt = 0; kt < 3; ++kt) ch = WMMA_BF16(ah[kt], ldb_frag_sw(sR, kt * 18 + j), ch);

    const int n = n0 + (lane & 15);
    const float bx = gb[n];
    const float bh = gb[288 + n];
    if (n0 < 192) {            // z and r columns: sum both sides now
#pragma unroll
      for (int i = 0; i < 8; ++i) {
        const int m = ((lane >> 4) << 3) + i;
        Szr[m * 192 + n] = (cx[i] + bx) + (ch[i] + bh);
      }
    } else {                   // candidate columns: keep sides separate
      const int q = n - 192;
#pragma unroll
      for (int i = 0; i < 8; ++i) {
        const int m = ((lane >> 4) << 3) + i;
        Sxh[m * 96 + q] = cx[i] + bx;
        Shh[m * 96 + q] = ch[i] + bh;
      }
    }
  }

  // elementwise GRU blend (per-wave LDS, wave-internal ordering via DScnt)
  for (int idx = lane; idx < 16 * 96; idx += 32) {
    const int m = idx / 96, k = idx - m * 96;
    const float z = 1.f / (1.f + __expf(-Szr[m * 192 + k]));
    const float r = 1.f / (1.f + __expf(-Szr[m * 192 + 96 + k]));
    const float cand = tanhf(Sxh[m * 96 + k] + r * Shh[m * 96 + k]);
    const size_t gi = (size_t)(node0 + m) * HID + k;
    const float hold = hf[gi];
    const float hnew = z * hold + (1.f - z) * cand;
    hf[gi] = hnew;
    hb[gi] = f2bf(hnew);
  }
}

// ---------------------------------------------------------------- readout
__global__ void out_kernel(const float* __restrict__ hf,
                           const float* __restrict__ Wout,
                           const float* __restrict__ bout,
                           float* __restrict__ out, int n_nodes) {
  const int node = blockIdx.x * blockDim.x + threadIdx.x;
  if (node >= n_nodes) return;
  const float* h = hf + (size_t)node * HID;
  float l[9];
#pragma unroll
  for (int j = 0; j < 9; ++j) l[j] = bout[j];
  for (int k = 0; k < HID; ++k) {
    const float hv = h[k];
#pragma unroll
    for (int j = 0; j < 9; ++j) l[j] += hv * Wout[k * 9 + j];
  }
  float mx = l[0];
#pragma unroll
  for (int j = 1; j < 9; ++j) mx = fmaxf(mx, l[j]);
  float s = 0.f;
#pragma unroll
  for (int j = 0; j < 9; ++j) { l[j] = __expf(l[j] - mx); s += l[j]; }
  const float inv = 1.f / s;
#pragma unroll
  for (int j = 0; j < 9; ++j) out[(size_t)node * 9 + j] = l[j] * inv;
}

// ---------------------------------------------------------------- host launch
extern "C" void kernel_launch(void* const* d_in, const int* in_sizes, int n_in,
                              void* d_out, int out_size, void* d_ws, size_t ws_size,
                              hipStream_t stream) {
  (void)in_sizes; (void)n_in; (void)out_size; (void)ws_size;
  const float* clues = (const float*)d_in[0];
  const float* h0    = (const float*)d_in[1];
  const int*   esrc  = (const int*)d_in[2];
  const int*   etgt  = (const int*)d_in[3];
  const float* W1    = (const float*)d_in[4];
  const float* b1    = (const float*)d_in[5];
  const float* W2    = (const float*)d_in[6];
  const float* b2    = (const float*)d_in[7];
  const float* W3    = (const float*)d_in[8];
  const float* b3    = (const float*)d_in[9];
  const float* gk    = (const float*)d_in[10];
  const float* grk   = (const float*)d_in[11];
  const float* gb    = (const float*)d_in[12];
  const float* Wout  = (const float*)d_in[13];
  const float* bout  = (const float*)d_in[14];
  float* out = (float*)d_out;

  // workspace layout (swizzled bf16 weights + state)
  char* ws = (char*)d_ws;
  unsigned short* W1s = (unsigned short*)ws;   ws += 36 * 1024;
  unsigned short* W2s = (unsigned short*)ws;   ws += 18 * 1024;
  unsigned short* W3s = (unsigned short*)ws;   ws += 18 * 1024;
  unsigned short* Ks  = (unsigned short*)ws;   ws += 72 * 1024;
  unsigned short* Rs  = (unsigned short*)ws;   ws += 54 * 1024;
  float*          hf  = (float*)ws;            ws += (size_t)N_NODES * HID * 4;
  unsigned short* hb  = (unsigned short*)ws;   ws += (size_t)N_NODES * HID * 2;
  float*          msgs = (float*)ws;           ws += (size_t)N_NODES * HID * 4;

  const int EDGE_SMEM = (36 + 18 + 18) * 1024 + 8 * (16 * 192 + 16 * 96) * 2; // 147456 B
  const int GRU_SMEM  = (72 + 54) * 1024 + 4 * 31744;                         // 256000 B
  (void)hipFuncSetAttribute((const void*)edge_mlp_kernel,
                            hipFuncAttributeMaxDynamicSharedMemorySize, EDGE_SMEM);
  (void)hipFuncSetAttribute((const void*)gru_kernel,
                            hipFuncAttributeMaxDynamicSharedMemorySize, GRU_SMEM);

  prep_weights_kernel<<<(72 * 512 + 255) / 256, 256, 0, stream>>>(
      W1, W2, W3, gk, grk, W1s, W2s, W3s, Ks, Rs);
  init_h_kernel<<<(N_NODES * HID + 255) / 256, 256, 0, stream>>>(h0, hf, hb, N_NODES * HID);

  for (int step = 0; step < NSTEPS; ++step) {
    (void)hipMemsetAsync(msgs, 0, (size_t)N_NODES * HID * 4, stream);
    edge_mlp_kernel<<<N_EDGES / 128, 256, EDGE_SMEM, stream>>>(
        hb, esrc, etgt, W1s, W2s, W3s, b1, b2, b3, msgs);
    gru_kernel<<<N_NODES / 64, 128, GRU_SMEM, stream>>>(
        clues, msgs, hf, hb, Ks, Rs, gb);
  }
  out_kernel<<<(N_NODES + 255) / 256, 256, 0, stream>>>(hf, Wout, bout, out, N_NODES);
}
